// MoEGate_7181185319145
// MI455X (gfx1250) — matrix operand order repaired and checked by reading, not compiled, not run on hardware
//
#include <hip/hip_runtime.h>
#include <hip/hip_bf16.h>
#include <math.h>

typedef __attribute__((ext_vector_type(16))) __bf16 v16bf;
typedef __attribute__((ext_vector_type(8)))  __bf16 v8bf;
typedef __attribute__((ext_vector_type(4)))  __bf16 v4bf;
typedef __attribute__((ext_vector_type(8)))  float  v8f;
typedef __attribute__((ext_vector_type(4)))  float  v4f;

#define NE      254     // real experts
#define NNULL   254     // null copies
#define TOPK    10
#define DDIM    2048
#define BT      16384   // B*T tokens
#define TPB     64      // tokens per block
#define KSTEP   64      // K panel per stage
#define LSTRIDE 257     // padded logits row stride (floats)

// ---------------- workspace layout (floats) ----------------
// [0..253]   P_real sums     [254..507] topk real counts
// [508]      null-pick count [509]      sum lse^2
__global__ __launch_bounds__(256) void moe_zero_ws(float* ws) {
  int i = blockIdx.x * blockDim.x + threadIdx.x;
  if (i < 2 * NE + 2) ws[i] = 0.f;
}

__global__ __launch_bounds__(256) void moe_gate_fused(
    const float* __restrict__ x, const float* __restrict__ gw,
    const float* __restrict__ bias, const float* __restrict__ nullp,
    float* __restrict__ out, float* __restrict__ ws)
{
  extern __shared__ char smem[];
  __bf16* As = (__bf16*)smem;               // 64 x 64 bf16  (staging, overlaid)
  __bf16* Bs = As + TPB * KSTEP;            // 256 x 64 bf16 (staging, overlaid)
  float*  Ls = (float*)smem;                // 64 x 257 f32 logits tile
  float*  Pblk = ((float*)smem) + TPB * LSTRIDE; // 254 f32 block-local P_real

  const int tid   = threadIdx.x;
  const int lane  = tid & 31;
  const int wave  = tid >> 5;
  const int mgrp  = wave & 3;    // which 16-token group
  const int nhalf = wave >> 2;   // which 128-expert half
  const int kh    = (lane >> 4) * 8;
  const long tok0 = (long)blockIdx.x * TPB;

  if (tid < NE) Pblk[tid] = 0.f;

  v8f acc[8] = {};

  for (int k0 = 0; k0 < DDIM; k0 += KSTEP) {
    // ---- stage A: 64x64 f32 -> bf16, float4 loads / bf16x4 stores ----
    #pragma unroll
    for (int j = 0; j < 4; ++j) {
      int q = tid + j * 256;            // quad index: 1024 quads
      int row = q >> 4, kq = (q & 15) * 4;
      v4f v = *(const v4f*)&x[(tok0 + row) * DDIM + k0 + kq];
      v4bf b;
      #pragma unroll
      for (int i = 0; i < 4; ++i) b[i] = (__bf16)v[i];
      *(v4bf*)&As[row * KSTEP + kq] = b;
    }
    // ---- stage B: 256x64, experts >= 254 padded with zeros ----
    #pragma unroll
    for (int j = 0; j < 16; ++j) {
      int q = tid + j * 256;            // quad index: 4096 quads
      int e = q >> 4, kq = (q & 15) * 4;
      v4f v = {};
      if (e < NE) v = *(const v4f*)&gw[(long)e * DDIM + k0 + kq];
      v4bf b;
      #pragma unroll
      for (int i = 0; i < 4; ++i) b[i] = (__bf16)v[i];
      *(v4bf*)&Bs[e * KSTEP + kq] = b;
    }
    if (k0 + KSTEP < DDIM) {
      __builtin_prefetch(&x[(tok0 + (tid >> 2)) * DDIM + k0 + KSTEP + (tid & 3) * 16], 0, 0);
    }
    __syncthreads();

    // ---- two K=32 halves of the panel ----
    #pragma unroll
    for (int h = 0; h < 2; ++h) {
      // A fragment: 16x32 bf16 (lane=M 0..15, K halves split across lane halves)
      const __bf16* arow = As + (mgrp * 16 + (lane & 15)) * KSTEP + h * 32;
      v8bf alo = *(const v8bf*)(arow + 2 * kh);
      v8bf ahi = *(const v8bf*)(arow + 2 * kh + 8);
      v16bf afrag;
      #pragma unroll
      for (int j = 0; j < 8; ++j) { afrag[j] = alo[j]; afrag[8 + j] = ahi[j]; }

      // preload ALL 8 B fragments, then issue WMMAs back-to-back
      v16bf bfrag[8];
      #pragma unroll
      for (int nt = 0; nt < 8; ++nt) {
        bfrag[nt] = *(const v16bf*)(Bs +
            ((nhalf * 8 + nt) * 16 + (lane & 15)) * KSTEP + h * 32 + (lane >> 4) * 16);
      }
      #pragma unroll
      for (int nt = 0; nt < 8; ++nt) {
        acc[nt] = __builtin_amdgcn_wmma_f32_16x16x32_bf16(
            false, afrag, false, bfrag[nt], (short)0, acc[nt], false, false);
      }
    }
    __syncthreads();
  }

  // ---- spill logits tile to LDS (C/D layout: row = r + 8*(lane>>4)) ----
  #pragma unroll
  for (int nt = 0; nt < 8; ++nt) {
    int col   = (nhalf * 8 + nt) * 16 + (lane & 15);
    int rbase = mgrp * 16 + (lane >> 4) * 8;
    #pragma unroll
    for (int r = 0; r < 8; ++r)
      Ls[(rbase + r) * LSTRIDE + col] = acc[nt][r];
  }
  __syncthreads();

  // ---------------- per-token epilogue ----------------
  const float nullv = nullp[0];
  if (tid < TPB) {
    const float* lrow = Ls + tid * LSTRIDE;

    // pass 1: real max + top-10 (desc, ties keep lowest index like jax top_k)
    float tv[TOPK]; int ti[TOPK];
    #pragma unroll
    for (int j = 0; j < TOPK; ++j) { tv[j] = -3.402823466e38f; ti[j] = -1; }
    float m = -3.402823466e38f;
    for (int e = 0; e < NE; ++e) {
      float l = lrow[e] + bias[e];
      m = fmaxf(m, l);
      if (l > tv[TOPK - 1]) {
        int j = TOPK - 1;
        while (j > 0 && tv[j - 1] < l) { tv[j] = tv[j - 1]; ti[j] = ti[j - 1]; --j; }
        tv[j] = l; ti[j] = e;
      }
    }
    float Mf = fmaxf(m, nullv);

    // pass 2: real-only softmax denom; full denom via rescale
    float s1 = 0.f;
    for (int e = 0; e < NE; ++e) s1 += __expf(lrow[e] + bias[e] - m);
    float sfull = s1 * __expf(m - Mf) + (float)NNULL * __expf(nullv - Mf);

    // pass 3: block-local P_real accumulation
    float inv_s1 = 1.f / s1;
    for (int e = 0; e < NE; ++e)
      atomicAdd(&Pblk[e], __expf(lrow[e] + bias[e] - m) * inv_s1);

    // merge nulls: reals with logit >= null win ties (lower index)
    int r = 0;
    #pragma unroll
    for (int j = 0; j < TOPK; ++j) r += (tv[j] >= nullv) ? 1 : 0;

    float invfull = 1.f / sfull;
    float wsum = 0.f;
    float pw[TOPK]; int pid[TOPK];
    #pragma unroll
    for (int j = 0; j < TOPK; ++j) {
      if (j < r) { pid[j] = ti[j]; pw[j] = __expf(tv[j] - Mf) * invfull; wsum += pw[j]; }
      else       { pid[j] = NE + (j - r); pw[j] = 0.f; }
    }
    wsum = fmaxf(wsum, 1e-6f);

    const long g = tok0 + tid;
    const long BTK = (long)BT * TOPK;
    for (int j = 0; j < TOPK; ++j) {
      int isn = (j >= r);
      out[g * TOPK + j]            = (float)pid[j];
      out[BTK + g * TOPK + j]      = isn ? 0.f : (pw[j] / wsum);
      out[2 * BTK + g * TOPK + j]  = (float)isn;
      if (!isn) atomicAdd(ws + NE + pid[j], 1.0f);
    }
    atomicAdd(ws + 2 * NE, (float)(TOPK - r));
    float lse = Mf + __logf(sfull);
    atomicAdd(ws + 2 * NE + 1, lse * lse);
  }
  __syncthreads();
  if (tid < NE) atomicAdd(ws + tid, Pblk[tid]);
}

__global__ void moe_finalize(const float* __restrict__ ws, float* __restrict__ out) {
  if (threadIdx.x == 0 && blockIdx.x == 0) {
    float csum = 0.f;
    for (int e = 0; e < NE; ++e) csum += ws[NE + e];
    float inv = 1.f / fmaxf(csum, 1e-6f);
    float lbal = 0.f;
    for (int e = 0; e < NE; ++e)
      lbal += (ws[NE + e] * inv) * (ws[e] * (1.f / (float)BT));
    lbal *= (float)NE;
    float nr = ws[2 * NE] / (float)((long)BT * TOPK);
    float lnull = (nr - 0.5f) * (nr - 0.5f);
    float lz = ws[2 * NE + 1] / (float)BT;
    out[(long)3 * BT * TOPK] = 0.02f * lbal + 0.001f * lz + 0.01f * lnull;
  }
}

extern "C" void kernel_launch(void* const* d_in, const int* in_sizes, int n_in,
                              void* d_out, int out_size, void* d_ws, size_t ws_size,
                              hipStream_t stream) {
  const float* x     = (const float*)d_in[0];
  const float* gw    = (const float*)d_in[1];
  const float* bias  = (const float*)d_in[2];
  const float* nullp = (const float*)d_in[3];
  float* out = (float*)d_out;
  float* ws  = (float*)d_ws;

  const size_t smem = (size_t)TPB * LSTRIDE * sizeof(float) + NE * sizeof(float);
  // >64KB dynamic LDS: legal on gfx1250 (320KB/WGP); raise the limit.
  (void)hipFuncSetAttribute((const void*)moe_gate_fused,
                            hipFuncAttributeMaxDynamicSharedMemorySize, (int)smem);

  moe_zero_ws<<<2, 256, 0, stream>>>(ws);
  moe_gate_fused<<<BT / TPB, 256, smem, stream>>>(x, gw, bias, nullp, out, ws);
  moe_finalize<<<1, 32, 0, stream>>>(ws, out);
}
